// WordAttention_25091198943702
// MI455X (gfx1250) — matrix operand order repaired and checked by reading, not compiled, not run on hardware
//
#include <hip/hip_runtime.h>
#include <math.h>
#include <stdint.h>

// ---------------------------------------------------------------------------
// WordAttention fused pipeline for gfx1250 (MI455X, wave32, WMMA).
// Dominant op: 4x [32768x512]x[512x512] GEMM + tanh + W2 reduction via
// v_wmma_f32_16x16x32_f16, with B-slabs double-buffered into LDS using
// GLOBAL_LOAD_ASYNC_TO_LDS_B128 (ASYNCcnt) so data movement overlaps WMMA.
// ---------------------------------------------------------------------------

typedef __attribute__((ext_vector_type(16))) _Float16 v16h;
typedef __attribute__((ext_vector_type(8)))  float    v8f;

union Frag16 {
    v16h h;
    unsigned int u[8];
};

#define BQ   64
#define CQ   512
#define WQ   512
#define AQ   4
#define OUTQ 512
#define NQ   (BQ * WQ)          // 32768 rows of xf
#define TEMP_INV 1.0f           // TEMP == 1.0

// ---------------------------------------------------------------------------
// Kernel 1: convert W1s (A*C*C f32) -> f16 in workspace.
// ---------------------------------------------------------------------------
__global__ void wa_cvt_w1(const float* __restrict__ w1, _Float16* __restrict__ out) {
    int i = blockIdx.x * blockDim.x + threadIdx.x;     // one float4 per thread
    float4 f = ((const float4*)w1)[i];
    union { _Float16 h[4]; uint2 v; } p;
    p.h[0] = (_Float16)f.x;
    p.h[1] = (_Float16)f.y;
    p.h[2] = (_Float16)f.z;
    p.h[3] = (_Float16)f.w;
    ((uint2*)out)[i] = p.v;
}

// ---------------------------------------------------------------------------
// Async copy of one contiguous 16KB B slab (16 rows of W1h) into LDS.
// Each of the 128 threads issues 8x async b128 copies (tracked by ASYNCcnt).
// LDS byte offset = low 32 bits of the flat shared-memory address (ISA 10.2).
// ---------------------------------------------------------------------------
__device__ __forceinline__ void stage_b_async(_Float16* __restrict__ dst,
                                              const _Float16* __restrict__ gsrc,
                                              int t) {
    #pragma unroll
    for (int i = 0; i < 8; ++i) {
        int idx = (t + i * 128) * 8;                      // 8 halves = 16 bytes
        unsigned           lo = (unsigned)(uintptr_t)(dst + idx);
        unsigned long long ga = (unsigned long long)(uintptr_t)(gsrc + idx);
        asm volatile("global_load_async_to_lds_b128 %0, %1, off"
                     :: "v"(lo), "v"(ga) : "memory");
    }
}

// ---------------------------------------------------------------------------
// Kernel 2: fused  s[n,a] = sum_o tanh( xf[n,:] . W1[a][o,:] ) * W2[a][o]
// Block: 128 threads = 4 waves; block owns 64 rows (each wave 16 rows).
// xf rows are just linear memory of x (C==W==512 makes the reshape trivial).
// A fragments live in 128 VGPRs per lane for the whole kernel; B slabs are
// shared by all 4 waves via double-buffered async LDS staging.
// ---------------------------------------------------------------------------
#define ROWS_PER_BLOCK 64
#define KTILES 16                 // 512 / 32
#define NPAIRS (AQ * (CQ / 16))   // 128 (a, o-tile) pairs

__launch_bounds__(128)
__global__ void wa_scores(const float* __restrict__ x,
                          const _Float16* __restrict__ w1h,
                          const float* __restrict__ w2,
                          float* __restrict__ s) {
    // 64 KB LDS: first used as the f16 A-tile, then (after the register
    // preload) re-used as two 16 KB ping-pong B-slab buffers.
    __shared__ _Float16 smem[ROWS_PER_BLOCK * CQ];

    const int t = threadIdx.x;

    // ---- stage 64 rows of x (f32, global, coalesced) -> f16 LDS ----
    const float4* xg = (const float4*)(x + (size_t)blockIdx.x * ROWS_PER_BLOCK * CQ);
    #pragma unroll 4
    for (int i = t; i < ROWS_PER_BLOCK * CQ / 4; i += 128) {
        float4 f = xg[i];
        union { _Float16 h[4]; uint2 v; } p;
        p.h[0] = (_Float16)f.x;
        p.h[1] = (_Float16)f.y;
        p.h[2] = (_Float16)f.z;
        p.h[3] = (_Float16)f.w;
        *(uint2*)&smem[i * 4] = p.v;
    }
    __syncthreads();

    const int wave = t >> 5;
    const int l    = t & 31;
    const int g    = l >> 4;       // lane group (0|1)
    const int ln   = l & 15;       // lane within group
    const int rowLocal = wave * 16;

    // ---- preload all A fragments for this wave's 16 rows (16 k-tiles) ----
    // A 16x32 f16 layout: lane = M, VGPR v: K = (v>=4)*16 + g*8 + 2*(v&3) (+pair)
    Frag16 afr[KTILES];
    const _Float16* arow = &smem[(rowLocal + ln) * CQ];
    #pragma unroll
    for (int kt = 0; kt < KTILES; ++kt) {
        #pragma unroll
        for (int v = 0; v < 8; ++v) {
            int kb = kt * 32 + ((v >= 4) ? 16 : 0) + g * 8 + (v & 3) * 2;
            afr[kt].u[v] = *(const unsigned int*)&arow[kb];
        }
    }
    __syncthreads();   // all waves done reading A-LDS; safe to reuse for B

    _Float16* bbuf0 = smem;                 // 16 KB
    _Float16* bbuf1 = smem + 16 * CQ;       // 16 KB

    const long rowGlobalBase = (long)blockIdx.x * ROWS_PER_BLOCK + rowLocal;

    // prologue: stage slab for pair 0
    stage_b_async(bbuf0, w1h, t);
    asm volatile("s_wait_asynccnt 0x0" ::: "memory");
    __syncthreads();

    float sacc[8];
    #pragma unroll
    for (int j = 0; j < 8; ++j) sacc[j] = 0.0f;

    for (int p = 0; p < NPAIRS; ++p) {
        const int a  = p >> 5;
        const int ot = p & 31;

        // issue async staging of the NEXT slab (overlaps with WMMA below)
        if (p + 1 < NPAIRS) {
            const int an  = (p + 1) >> 5;
            const int otn = (p + 1) & 31;
            const _Float16* gsrc = w1h + ((size_t)an * CQ + (size_t)otn * 16) * CQ;
            stage_b_async(((p + 1) & 1) ? bbuf1 : bbuf0, gsrc, t);
        }

        // ---- 16-WMMA K-chain on the current slab (two independent chains) ----
        // B 32x16 f16 layout: lane = N (=o), VGPR v: K = (v>=4)*16 + g*8 + 2*(v&3) (+pair)
        const _Float16* brow = ((p & 1) ? bbuf1 : bbuf0) + ln * CQ + g * 8;
        v8f acc0 = {}, acc1 = {};
        #pragma unroll
        for (int kt = 0; kt < KTILES; kt += 2) {
            Frag16 bf0, bf1;
            #pragma unroll
            for (int v = 0; v < 8; ++v) {
                bf0.u[v] = *(const unsigned int*)&brow[kt * 32 + ((v >= 4) ? 16 : 0) + (v & 3) * 2];
                bf1.u[v] = *(const unsigned int*)&brow[(kt + 1) * 32 + ((v >= 4) ? 16 : 0) + (v & 3) * 2];
            }
            acc0 = __builtin_amdgcn_wmma_f32_16x16x32_f16(
                     false, afr[kt].h,     false, bf0.h, (short)0, acc0, false, false);
            acc1 = __builtin_amdgcn_wmma_f32_16x16x32_f16(
                     false, afr[kt + 1].h, false, bf1.h, (short)0, acc1, false, false);
        }

        // epilogue: tanh + dot with W2 column slice.
        // C/D layout: VGPR j -> row m = j + 8*g, col o = ot*16 + ln
        const float w2v = w2[a * CQ + ot * 16 + ln];
        #pragma unroll
        for (int j = 0; j < 8; ++j)
            sacc[j] += tanhf(acc0[j] + acc1[j]) * w2v;

        if (ot == 31) {
            // reduce the 16 lanes of each group (they hold different o-columns)
            #pragma unroll
            for (int j = 0; j < 8; ++j) {
                float v = sacc[j];
                #pragma unroll
                for (int m = 1; m < 16; m <<= 1)
                    v += __shfl_xor(v, m, 32);
                if (ln == 0) {
                    long n = rowGlobalBase + g * 8 + j;
                    s[n * AQ + a] = v;
                }
                sacc[j] = 0.0f;
            }
        }

        // next slab's async copy must have landed; everyone done reading cur
        asm volatile("s_wait_asynccnt 0x0" ::: "memory");
        __syncthreads();
    }
}

// ---------------------------------------------------------------------------
// Kernel 3: per-batch softmax over W for each head, then sum heads -> weight.
// scores[b,a,w] = s[(b*W + w)*A + a];  weight written into d_out (2nd half).
// ---------------------------------------------------------------------------
__launch_bounds__(256)
__global__ void wa_softmax(const float* __restrict__ s,
                           const float* __restrict__ mask,
                           float* __restrict__ weight) {
    __shared__ float red[256];
    const int b = blockIdx.x;
    const int t = threadIdx.x;

    const float m0 = mask[b * WQ + t];
    const float m1 = mask[b * WQ + t + 256];
    float wsum0 = 0.0f, wsum1 = 0.0f;

    for (int a = 0; a < AQ; ++a) {
        float v0 = (s[((size_t)b * WQ + t)       * AQ + a] + m0) * TEMP_INV;
        float v1 = (s[((size_t)b * WQ + t + 256) * AQ + a] + m1) * TEMP_INV;

        red[t] = fmaxf(v0, v1);
        __syncthreads();
        for (int st = 128; st > 0; st >>= 1) {
            if (t < st) red[t] = fmaxf(red[t], red[t + st]);
            __syncthreads();
        }
        float mx = red[0];
        __syncthreads();

        float e0 = __expf(v0 - mx);
        float e1 = __expf(v1 - mx);
        red[t] = e0 + e1;
        __syncthreads();
        for (int st = 128; st > 0; st >>= 1) {
            if (t < st) red[t] += red[t + st];
            __syncthreads();
        }
        float inv = 1.0f / red[0];
        __syncthreads();

        wsum0 += e0 * inv;
        wsum1 += e1 * inv;
    }
    weight[b * WQ + t]       = wsum0;
    weight[b * WQ + t + 256] = wsum1;
}

// ---------------------------------------------------------------------------
// Kernel 4: pooled[b,c] = max_w x[b,c,0,w] * weight[b,w]
// ---------------------------------------------------------------------------
__launch_bounds__(256)
__global__ void wa_pool(const float* __restrict__ x,
                        const float* __restrict__ weight,
                        float* __restrict__ pooled) {
    __shared__ float wsm[WQ];
    const int b = blockIdx.x;
    const int t = threadIdx.x;
    wsm[t]       = weight[b * WQ + t];
    wsm[t + 256] = weight[b * WQ + t + 256];
    __syncthreads();

    for (int c = t; c < CQ; c += 256) {
        const float* xr = x + ((size_t)b * CQ + c) * WQ;
        float mx = -__builtin_huge_valf();
        for (int w = 0; w < WQ; ++w)
            mx = fmaxf(mx, xr[w] * wsm[w]);
        pooled[b * CQ + c] = mx;
    }
}

// ---------------------------------------------------------------------------
// Kernel 5: out[b,o] = pooled[b,:] . Wm[o,:]   (tiny: 34 MFLOP, full f32)
// ---------------------------------------------------------------------------
__launch_bounds__(256)
__global__ void wa_out(const float* __restrict__ pooled,
                       const float* __restrict__ wm,
                       float* __restrict__ out) {
    const int gid = blockIdx.x * 256 + threadIdx.x;   // = b*512 + o
    const int b = gid >> 9;
    const int o = gid & 511;
    const float* p = pooled + b * CQ;
    const float* w = wm + (size_t)o * CQ;
    float acc = 0.0f;
    #pragma unroll 8
    for (int c = 0; c < CQ; ++c)
        acc = fmaf(p[c], w[c], acc);
    out[gid] = acc;
}

// ---------------------------------------------------------------------------
// Host-side launcher.
// d_in: 0=x [B,C,1,W] f32, 1=mask [B,1,1,W] f32, 2=W1s [A,C,C] f32,
//       3=W2s [A,1,C] f32, 4=Wm [OUT,C] f32
// d_out: out (B*OUT floats) then weight (B*W floats)
// ---------------------------------------------------------------------------
extern "C" void kernel_launch(void* const* d_in, const int* in_sizes, int n_in,
                              void* d_out, int out_size, void* d_ws, size_t ws_size,
                              hipStream_t stream) {
    (void)in_sizes; (void)n_in; (void)out_size; (void)ws_size;

    const float* x    = (const float*)d_in[0];
    const float* mask = (const float*)d_in[1];
    const float* W1s  = (const float*)d_in[2];
    const float* W2s  = (const float*)d_in[3];
    const float* Wm   = (const float*)d_in[4];

    float* out    = (float*)d_out;           // [B*OUT]
    float* weight = out + BQ * OUTQ;         // [B*W]

    char* ws = (char*)d_ws;
    float*    s      = (float*)ws;                                     // N*A f32 (512 KB)
    float*    pooled = (float*)(ws + (size_t)NQ * AQ * sizeof(float)); // B*C f32 (128 KB)
    _Float16* w1h    = (_Float16*)(ws + (size_t)NQ * AQ * sizeof(float)
                                      + (size_t)BQ * CQ * sizeof(float)); // A*C*C f16 (2 MB)

    // 1) W1 f32 -> f16  (A*C*C/4 float4s)
    wa_cvt_w1<<<(AQ * CQ * CQ / 4) / 256, 256, 0, stream>>>(W1s, w1h);

    // 2) fused per-head GEMM + tanh + W2 reduction -> scores
    wa_scores<<<NQ / ROWS_PER_BLOCK, 128, 0, stream>>>(x, w1h, W2s, s);

    // 3) softmax over W per head, head-sum -> weight (d_out second half)
    wa_softmax<<<BQ, 256, 0, stream>>>(s, mask, weight);

    // 4) weighted max-pool over W
    wa_pool<<<BQ, 256, 0, stream>>>(x, weight, pooled);

    // 5) final projection (f32)
    wa_out<<<(BQ * OUTQ) / 256, 256, 0, stream>>>(pooled, Wm, out);
}
